// cross_attention_block_78262894068462
// MI455X (gfx1250) — matrix operand order repaired and checked by reading, not compiled, not run on hardware
//
#include <hip/hip_runtime.h>
#include <hip/hip_bf16.h>

// ---------------------------------------------------------------------------
// Swin-3D cross-attention block for MI455X (gfx1250, wave32, WMMA).
// All GEMM-shaped math -> v_wmma_f32_16x16x32_f16 (f16 in, f32 accum).
// Attention operand staging -> GLOBAL_LOAD_ASYNC_TO_LDS_B128 (ASYNCcnt).
// Row-wise math (LN, softmax, residual, conv) stays fp32 VALU.
// ---------------------------------------------------------------------------

typedef _Float16 h8   __attribute__((ext_vector_type(8)));
typedef _Float16 v16h __attribute__((ext_vector_type(16)));
typedef float    v8f  __attribute__((ext_vector_type(8)));
typedef int      ai4  __attribute__((vector_size(16)));   // async-copy payload

#define NWIN   1024      // 2 * 8 * 8 * 8 windows
#define NTOK   150       // 5*6*5 tokens per window
#define NPAD   160       // padded to multiple of 16
#define TPAD   (NWIN*NPAD)
#define NTOT   153600    // 2*40*48*40 valid tokens
#define CH     96

// --- gfx1250 async global->LDS support (guarded; falls back to VGPR copy) ---
#if defined(__has_builtin)
#if __has_builtin(__builtin_amdgcn_global_load_async_to_lds_b128)
#define HAVE_ASYNC_LDS 1
#endif
#endif
#ifndef HAVE_ASYNC_LDS
#define HAVE_ASYNC_LDS 0
#endif

#if defined(__has_builtin)
#if __has_builtin(__builtin_amdgcn_s_wait_asynccnt)
#define ASYNC_WAIT() __builtin_amdgcn_s_wait_asynccnt(0)
#endif
#endif
#ifndef ASYNC_WAIT
#define ASYNC_WAIT() asm volatile("s_wait_asynccnt 0" ::: "memory")
#endif

__device__ __forceinline__ v16h frag_combine(h8 lo, h8 hi) {
  v16h r;
#pragma unroll
  for (int i = 0; i < 8; ++i) { r[i] = lo[i]; r[8 + i] = hi[i]; }
  return r;
}

// A-fragment (16x32 f16, row-major source, row stride ld halves).
// ISA layout: lanes 0-15 row M=lane, K elements {0..7,16..23}; lanes 16-31
// same rows, K {8..15,24..31}.  => two contiguous 16B chunks per lane.
__device__ __forceinline__ v16h load_a(const _Float16* base, int ld, int m0, int k0) {
  int lane = threadIdx.x & 31;
  int row  = m0 + (lane & 15);
  int ko   = k0 + ((lane >> 4) << 3);            // +8 for upper half-wave
  const _Float16* p = base + (size_t)row * ld + ko;
  h8 lo = *(const h8*)p;
  h8 hi = *(const h8*)(p + 16);
  return frag_combine(lo, hi);
}

// B-fragment (32x16 f16). Source is stored column-transposed: base[n*ld + k]
// holds B[k][n]. Lanes 0-15: column n=lane, K={0..7,16..23}; lanes 16-31: K={8..15,24..31}.
__device__ __forceinline__ v16h load_b(const _Float16* base, int ld, int n0, int k0) {
  int lane = threadIdx.x & 31;
  int n    = n0 + (lane & 15);
  int kb   = k0 + ((lane >> 4) << 3);            // +8 for upper half-wave
  const _Float16* p = base + (size_t)n * ld + kb;
  h8 lo = *(const h8*)p;
  h8 hi = *(const h8*)(p + 16);
  return frag_combine(lo, hi);
}

__device__ __forceinline__ v8f wmma_f16(v16h a, v16h b, v8f c) {
  return __builtin_amdgcn_wmma_f32_16x16x32_f16(false, a, false, b, (short)0, c,
                                                false, false);
}

__device__ __forceinline__ float gelu_tanh(float x) {
  float x3 = x * x * x;
  float t  = tanhf(0.7978845608028654f * (x + 0.044715f * x3));
  return 0.5f * x * (1.0f + t);
}

// ---------------------------------------------------------------------------
// K1: LayerNorm + window partition (fp32 NCDHW -> f16 [win][160][96], padded
// rows zero-filled).
// ---------------------------------------------------------------------------
__global__ void k_ln_partition(const float* __restrict__ Im, const float* __restrict__ If,
                               const float* g1q, const float* b1q,
                               const float* g1kv, const float* b1kv,
                               _Float16* __restrict__ xw, _Float16* __restrict__ yw) {
  int p = blockIdx.x * blockDim.x + threadIdx.x;
  if (p >= TPAD) return;
  int win = p / NPAD, r = p % NPAD;
  _Float16* xo = xw + (size_t)p * CH;
  _Float16* yo = yw + (size_t)p * CH;
  if (r >= NTOK) {
    for (int c = 0; c < CH; ++c) { xo[c] = (_Float16)0.f; yo[c] = (_Float16)0.f; }
    return;
  }
  int b = win >> 9, rem = win & 511;
  int db = rem >> 6, hb = (rem >> 3) & 7, wb = rem & 7;
  int rd = r / 30, rh = (r / 5) % 6, rw = r % 5;
  int d = db * 5 + rd, h = hb * 6 + rh, w = wb * 5 + rw;
  size_t base = (size_t)b * CH * 76800 + (size_t)d * 1920 + (size_t)h * 40 + w;

  float s = 0.f, s2 = 0.f;
  for (int c = 0; c < CH; ++c) { float v = Im[base + (size_t)c * 76800]; s += v; s2 += v * v; }
  float m = s * (1.f / CH), var = s2 * (1.f / CH) - m * m;
  float rs = rsqrtf(var + 1e-5f);
  for (int c = 0; c < CH; ++c) {
    float v = Im[base + (size_t)c * 76800];
    xo[c] = (_Float16)(((v - m) * rs) * g1q[c] + b1q[c]);
  }
  s = 0.f; s2 = 0.f;
  for (int c = 0; c < CH; ++c) { float v = If[base + (size_t)c * 76800]; s += v; s2 += v * v; }
  m = s * (1.f / CH); var = s2 * (1.f / CH) - m * m; rs = rsqrtf(var + 1e-5f);
  for (int c = 0; c < CH; ++c) {
    float v = If[base + (size_t)c * 76800];
    yo[c] = (_Float16)(((v - m) * rs) * g1kv[c] + b1kv[c]);
  }
}

// ---------------------------------------------------------------------------
// K2: flat GEMM [TPAD x 96] @ [96 x Nout] + bias -> f16.  Weight transposed
// into LDS for contiguous B-fragment reads.  One wave per 16-row M-tile.
// Nout==192 splits columns into out0 (K) / out1 (V).
// ---------------------------------------------------------------------------
__global__ void k_gemm_f16(const _Float16* __restrict__ A, const float* __restrict__ Wsrc,
                           const float* __restrict__ bias,
                           _Float16* __restrict__ out0, _Float16* __restrict__ out1,
                           int Nout) {
  __shared__ _Float16 Wt[192 * CH];           // [Nout][96] transposed weights
  int tid = threadIdx.x;
  int tot = Nout * CH;
  for (int e = tid; e < tot; e += 256) {
    int n = e / CH, k = e % CH;
    Wt[n * CH + k] = (_Float16)Wsrc[k * Nout + n];
  }
  __syncthreads();
  int lane = tid & 31, wave = tid >> 5;
  int m0 = (blockIdx.x * 8 + wave) * 16;
  v16h a0 = load_a(A, CH, m0, 0);
  v16h a1 = load_a(A, CH, m0, 32);
  v16h a2 = load_a(A, CH, m0, 64);
  int ntiles = Nout >> 4;
  for (int nt = 0; nt < ntiles; ++nt) {
    v8f acc = {};
    acc = wmma_f16(a0, load_b(Wt, CH, nt * 16, 0),  acc);
    acc = wmma_f16(a1, load_b(Wt, CH, nt * 16, 32), acc);
    acc = wmma_f16(a2, load_b(Wt, CH, nt * 16, 64), acc);
    int col = nt * 16 + (lane & 15);
    float bv = bias[col];
#pragma unroll
    for (int i = 0; i < 8; ++i) {
      int row = m0 + ((lane < 16) ? i : i + 8);
      float v = acc[i] + bv;
      if (col < CH) out0[(size_t)row * CH + col] = (_Float16)v;
      else          out1[(size_t)row * CH + (col - CH)] = (_Float16)v;
    }
  }
}

// ---------------------------------------------------------------------------
// K3: per-window attention. One workgroup (8 waves) per window.
// Q/K/V staged via async global->LDS b128 (ASYNCcnt); V transposed LDS->LDS.
// S (fp32) and P (f16) in LDS; fp32 softmax with on-the-fly rel-pos bias.
// ---------------------------------------------------------------------------
__global__ void k_attn(const _Float16* __restrict__ Qg, const _Float16* __restrict__ Kg,
                       const _Float16* __restrict__ Vg, const float* __restrict__ relb,
                       _Float16* __restrict__ Og) {
  extern __shared__ __align__(16) char smem[];
  _Float16* Qs  = (_Float16*)smem;               // 160*96
  _Float16* Ks  = Qs + NPAD * CH;                // 160*96
  _Float16* Vts = Ks + NPAD * CH;                // [96][160]
  float*    Ss  = (float*)(Vts + NPAD * CH);     // 160*160 fp32
  _Float16* Ps  = (_Float16*)(Ss + NPAD * NPAD); // 160*160 f16 (also V staging)
  float*    bS  = (float*)(Ps + NPAD * NPAD);    // 891 fp32

  int tid = threadIdx.x, lane = tid & 31, wave = tid >> 5;
  int win = blockIdx.x;
  size_t wbase = (size_t)win * NPAD * CH;

  // --- stage Q, K row-major; V row-major into Ps scratch (transposed later).
#if HAVE_ASYNC_LDS
  for (int e = tid; e < 3 * (NPAD * CH / 8); e += 256) {
    int mtx = e / (NPAD * CH / 8), idx = e % (NPAD * CH / 8);
    const _Float16* gsrc =
        (mtx == 0 ? Qg : mtx == 1 ? Kg : Vg) + wbase + (size_t)idx * 8;
    _Float16* ldst = (mtx == 0 ? Qs : mtx == 1 ? Ks : Ps) + idx * 8;
    __builtin_amdgcn_global_load_async_to_lds_b128(
        (__attribute__((address_space(1))) ai4*)(void*)gsrc,
        (__attribute__((address_space(3))) ai4*)(void*)ldst, 0, 0);
  }
  ASYNC_WAIT();
#else
  {
    const uint2* qsrc = (const uint2*)(Qg + wbase);
    const uint2* ksrc = (const uint2*)(Kg + wbase);
    const uint2* vsrc = (const uint2*)(Vg + wbase);
    uint2* qdst = (uint2*)Qs; uint2* kdst = (uint2*)Ks; uint2* vdst = (uint2*)Ps;
    for (int e = tid; e < NPAD * CH / 4; e += 256) {
      qdst[e] = qsrc[e]; kdst[e] = ksrc[e]; vdst[e] = vsrc[e];
    }
  }
#endif
  for (int e = tid; e < 891; e += 256) bS[e] = relb[e];
  __syncthreads();

  // V transpose: Ps staging [token][chan] -> Vts [chan][token] (LDS->LDS).
  for (int e = tid; e < NPAD * CH; e += 256) {
    int r = e / CH, c = e % CH;
    Vts[c * NPAD + r] = Ps[e];
  }
  __syncthreads();

  // S = Q @ K^T  (100 16x16 tiles over 8 waves, K-dim = 96 channels)
  for (int t = wave; t < 100; t += 8) {
    int mt = t / 10, nt = t % 10;
    v8f acc = {};
#pragma unroll
    for (int kb = 0; kb < 3; ++kb)
      acc = wmma_f16(load_a(Qs, CH, mt * 16, kb * 32),
                     load_b(Ks, CH, nt * 16, kb * 32), acc);
    int col = nt * 16 + (lane & 15);
#pragma unroll
    for (int i = 0; i < 8; ++i) {
      int row = mt * 16 + ((lane < 16) ? i : i + 8);
      Ss[row * NPAD + col] = acc[i];
    }
  }
  __syncthreads();

  // fp32 softmax per query row, with Swin relative-position bias on the fly.
  const float scale = 0.1020620726159658f;   // 96^-0.5
  if (tid < NPAD) {
    int r = tid;
    if (r < NTOK) {
      int rd = r / 30, rh = (r / 5) % 6, rw = r % 5;
      float mx = -1e30f;
      int m = 0;
      for (int md = 0; md < 5; ++md)
        for (int mh = 0; mh < 6; ++mh)
          for (int mw = 0; mw < 5; ++mw) {
            int bidx = (rd - md + 4) * 99 + (rh - mh + 5) * 9 + (rw - mw + 4);
            float v = Ss[r * NPAD + m] * scale + bS[bidx];
            Ss[r * NPAD + m] = v;
            if (v > mx) mx = v;
            ++m;
          }
      float sum = 0.f;
      for (m = 0; m < NTOK; ++m) {
        float e = __expf(Ss[r * NPAD + m] - mx);
        Ss[r * NPAD + m] = e; sum += e;
      }
      float rinv = 1.f / sum;
      for (m = 0; m < NTOK; ++m) Ps[r * NPAD + m] = (_Float16)(Ss[r * NPAD + m] * rinv);
      for (m = NTOK; m < NPAD; ++m) Ps[r * NPAD + m] = (_Float16)0.f;
    } else {
      for (int m = 0; m < NPAD; ++m) Ps[r * NPAD + m] = (_Float16)0.f;
    }
  }
  __syncthreads();

  // O = P @ V   (60 tiles, K-dim = 160 padded key tokens)
  for (int t = wave; t < 60; t += 8) {
    int mt = t / 6, nt = t % 6;
    v8f acc = {};
#pragma unroll
    for (int kb = 0; kb < 5; ++kb)
      acc = wmma_f16(load_a(Ps, NPAD, mt * 16, kb * 32),
                     load_b(Vts, NPAD, nt * 16, kb * 32), acc);
    int col = nt * 16 + (lane & 15);
#pragma unroll
    for (int i = 0; i < 8; ++i) {
      int row = mt * 16 + ((lane < 16) ? i : i + 8);
      Og[wbase + (size_t)row * CH + col] = (_Float16)acc[i];
    }
  }
}

// ---------------------------------------------------------------------------
// K4: projection GEMM [TPAD x 96]@[96x96] + bias -> fp32 (windowed layout).
// ---------------------------------------------------------------------------
__global__ void k_gemm_proj(const _Float16* __restrict__ A, const float* __restrict__ Wsrc,
                            const float* __restrict__ bias, float* __restrict__ out) {
  __shared__ _Float16 Wt[CH * CH];
  int tid = threadIdx.x;
  for (int e = tid; e < CH * CH; e += 256) {
    int n = e / CH, k = e % CH;
    Wt[n * CH + k] = (_Float16)Wsrc[k * CH + n];
  }
  __syncthreads();
  int lane = tid & 31, wave = tid >> 5;
  int m0 = (blockIdx.x * 8 + wave) * 16;
  v16h a0 = load_a(A, CH, m0, 0);
  v16h a1 = load_a(A, CH, m0, 32);
  v16h a2 = load_a(A, CH, m0, 64);
  for (int nt = 0; nt < 6; ++nt) {
    v8f acc = {};
    acc = wmma_f16(a0, load_b(Wt, CH, nt * 16, 0),  acc);
    acc = wmma_f16(a1, load_b(Wt, CH, nt * 16, 32), acc);
    acc = wmma_f16(a2, load_b(Wt, CH, nt * 16, 64), acc);
    int col = nt * 16 + (lane & 15);
    float bv = bias[col];
#pragma unroll
    for (int i = 0; i < 8; ++i) {
      int row = m0 + ((lane < 16) ? i : i + 8);
      out[(size_t)row * CH + col] = acc[i] + bv;
    }
  }
}

// ---------------------------------------------------------------------------
// K5: window-reverse + residual + LayerNorm2.
// X (fp32, NDHWC) = shortcut + proj;  h (f16, NDHWC) = LN2(X).
// ---------------------------------------------------------------------------
__global__ void k_residual_ln(const float* __restrict__ Im, const float* __restrict__ PR,
                              const float* g2, const float* b2,
                              float* __restrict__ Xb, _Float16* __restrict__ hb) {
  int t = blockIdx.x * blockDim.x + threadIdx.x;
  if (t >= NTOT) return;
  int win = t / NTOK, r = t % NTOK;
  size_t p = (size_t)win * NPAD + r;
  int b = win >> 9, rem = win & 511;
  int db = rem >> 6, hbk = (rem >> 3) & 7, wbk = rem & 7;
  int rd = r / 30, rh = (r / 5) % 6, rw = r % 5;
  int d = db * 5 + rd, h = hbk * 6 + rh, w = wbk * 5 + rw;
  size_t sflat = ((((size_t)b * 40 + d) * 48 + h) * 40 + w);
  size_t ibase = (size_t)b * CH * 76800 + (size_t)d * 1920 + (size_t)h * 40 + w;
  const float* pr = PR + p * CH;

  float s = 0.f, s2 = 0.f;
  for (int c = 0; c < CH; ++c) {
    float v = Im[ibase + (size_t)c * 76800] + pr[c];
    s += v; s2 += v * v;
  }
  float m = s * (1.f / CH), var = s2 * (1.f / CH) - m * m;
  float rs = rsqrtf(var + 1e-5f);
  float* xo = Xb + sflat * CH;
  _Float16* ho = hb + sflat * CH;
  for (int c = 0; c < CH; ++c) {
    float v = Im[ibase + (size_t)c * 76800] + pr[c];
    xo[c] = v;
    ho[c] = (_Float16)(((v - m) * rs) * g2[c] + b2[c]);
  }
}

// ---------------------------------------------------------------------------
// K6: fused MLP: gelu(h @ W1 + b1) @ W2 + b2 + X -> x2 (fp32).
// 128 rows / workgroup, both transposed weights + hidden tile in LDS.
// ---------------------------------------------------------------------------
__global__ void k_mlp(const _Float16* __restrict__ hb, const float* __restrict__ Xb,
                      const float* __restrict__ w1, const float* __restrict__ b1,
                      const float* __restrict__ w2, const float* __restrict__ b2,
                      float* __restrict__ x2) {
  extern __shared__ __align__(16) char smem[];
  _Float16* W1t = (_Float16*)smem;           // [384][96]
  _Float16* W2t = W1t + 384 * CH;            // [96][384]
  _Float16* ht  = W2t + 384 * CH;            // [128][384] hidden tile
  float* b1s = (float*)(ht + 128 * 384);
  float* b2s = b1s + 384;

  int tid = threadIdx.x, lane = tid & 31, wave = tid >> 5;
  for (int e = tid; e < 384 * CH; e += 256) {
    int n = e / CH, k = e % CH;
    W1t[n * CH + k] = (_Float16)w1[k * 384 + n];
  }
  for (int e = tid; e < CH * 384; e += 256) {
    int n = e / 384, k = e % 384;
    W2t[n * 384 + k] = (_Float16)w2[k * CH + n];
  }
  for (int e = tid; e < 384; e += 256) b1s[e] = b1[e];
  if (tid < CH) b2s[tid] = b2[tid];
  __syncthreads();

  int m0 = blockIdx.x * 128 + wave * 16;
  // Warm L2/L0 for the phase-2 fp32 residual reads (global_prefetch_b8).
  __builtin_prefetch(Xb + (size_t)(m0 + (lane & 15)) * CH, 0, 3);

  v16h a0 = load_a(hb, CH, m0, 0);
  v16h a1 = load_a(hb, CH, m0, 32);
  v16h a2 = load_a(hb, CH, m0, 64);
  for (int nt = 0; nt < 24; ++nt) {
    v8f acc = {};
    acc = wmma_f16(a0, load_b(W1t, CH, nt * 16, 0),  acc);
    acc = wmma_f16(a1, load_b(W1t, CH, nt * 16, 32), acc);
    acc = wmma_f16(a2, load_b(W1t, CH, nt * 16, 64), acc);
    int col = nt * 16 + (lane & 15);
    float bv = b1s[col];
#pragma unroll
    for (int i = 0; i < 8; ++i) {
      int row = wave * 16 + ((lane < 16) ? i : i + 8);
      ht[row * 384 + col] = (_Float16)gelu_tanh(acc[i] + bv);
    }
  }
  __syncthreads();

  for (int nt = 0; nt < 6; ++nt) {
    v8f acc = {};
#pragma unroll
    for (int kb = 0; kb < 12; ++kb)
      acc = wmma_f16(load_a(ht, 384, wave * 16, kb * 32),
                     load_b(W2t, 384, nt * 16, kb * 32), acc);
    int col = nt * 16 + (lane & 15);
    float bv = b2s[col];
#pragma unroll
    for (int i = 0; i < 8; ++i) {
      int row = m0 + ((lane < 16) ? i : i + 8);
      size_t ad = (size_t)row * CH + col;
      x2[ad] = acc[i] + bv + Xb[ad];
    }
  }
}

// ---------------------------------------------------------------------------
// K7: 3x3x3 conv, 96 -> 3 channels (flow head), NDHWC input, NCDHW output.
// ---------------------------------------------------------------------------
__global__ void k_conv(const float* __restrict__ x2, const float* __restrict__ cw,
                       const float* __restrict__ cb, float* __restrict__ out) {
  __shared__ float wS[3 * 27 * CH];  // [oc][kk][c]
  __shared__ float cbS[3];
  int tid = threadIdx.x;
  for (int e = tid; e < 3 * 27 * CH; e += 256) {
    int oc = e / (27 * CH);
    int rem = e % (27 * CH);
    int kk = rem / CH, c = rem % CH;
    wS[e] = cw[((size_t)oc * CH + c) * 27 + kk];
  }
  if (tid < 3) cbS[tid] = cb[tid];
  __syncthreads();

  int g = blockIdx.x * blockDim.x + tid;
  if (g >= 460800) return;
  int w = g % 40;
  int h = (g / 40) % 48;
  int d = (g / 1920) % 40;
  int oc = (g / 76800) % 3;
  int b = g / 230400;
  float acc = cbS[oc];
  for (int kd = 0; kd < 3; ++kd) {
    int dd = d + kd - 1; if (dd < 0 || dd >= 40) continue;
    for (int kh = 0; kh < 3; ++kh) {
      int hh = h + kh - 1; if (hh < 0 || hh >= 48) continue;
      for (int kw = 0; kw < 3; ++kw) {
        int ww = w + kw - 1; if (ww < 0 || ww >= 40) continue;
        const float* xp = x2 + ((((size_t)b * 40 + dd) * 48 + hh) * 40 + ww) * CH;
        const float* wp = wS + (oc * 27 + (kd * 9 + kh * 3 + kw)) * CH;
        float a = 0.f;
        for (int c = 0; c < CH; ++c) a += xp[c] * wp[c];
        acc += a;
      }
    }
  }
  out[g] = acc;
}

// ---------------------------------------------------------------------------
extern "C" void kernel_launch(void* const* d_in, const int* in_sizes, int n_in,
                              void* d_out, int out_size, void* d_ws, size_t ws_size,
                              hipStream_t stream) {
  (void)in_sizes; (void)n_in; (void)out_size; (void)ws_size;
  const float* Im     = (const float*)d_in[0];
  const float* If     = (const float*)d_in[1];
  const float* ln1q_g = (const float*)d_in[2];
  const float* ln1q_b = (const float*)d_in[3];
  const float* ln1kv_g= (const float*)d_in[4];
  const float* ln1kv_b= (const float*)d_in[5];
  const float* wq     = (const float*)d_in[6];
  const float* bq     = (const float*)d_in[7];
  const float* wkv    = (const float*)d_in[8];
  const float* bkv    = (const float*)d_in[9];
  const float* wproj  = (const float*)d_in[10];
  const float* bproj  = (const float*)d_in[11];
  const float* relb   = (const float*)d_in[12];
  const float* ln2_g  = (const float*)d_in[13];
  const float* ln2_b  = (const float*)d_in[14];
  const float* mlp_w1 = (const float*)d_in[15];
  const float* mlp_b1 = (const float*)d_in[16];
  const float* mlp_w2 = (const float*)d_in[17];
  const float* mlp_b2 = (const float*)d_in[18];
  const float* conv_w = (const float*)d_in[19];
  const float* conv_b = (const float*)d_in[20];

  // Workspace layout (bytes), with lifetime-based region reuse:
  //   seg0: xw f16  (later Ob)
  //   seg1+seg2: Qb,Kb f16 (later PR fp32, exactly 2 segs)
  //   seg3+seg4: yw,Vb f16 (later Xb fp32, 59MB <= 62.9MB)
  //   seg5: hb f16 ; seg6: x2 fp32.  Total ~245.8 MB.
  char* ws = (char*)d_ws;
  const size_t S16 = (size_t)TPAD * CH * 2;        // 31,457,280
  _Float16* xw = (_Float16*)(ws);
  _Float16* Qb = (_Float16*)(ws + S16);
  _Float16* Kb = (_Float16*)(ws + 2 * S16);
  _Float16* yw = (_Float16*)(ws + 3 * S16);
  _Float16* Vb = (_Float16*)(ws + 4 * S16);
  _Float16* Ob = xw;                                // reuse seg0
  float*    PR = (float*)(ws + S16);                // reuse seg1+seg2
  float*    Xb = (float*)(ws + 3 * S16);            // reuse seg3+seg4
  _Float16* hb = (_Float16*)(ws + 5 * S16);
  float*    x2 = (float*)(ws + 5 * S16 + (size_t)NTOT * CH * 2);

  const int ATTN_SMEM = NPAD * CH * 2 * 3 + NPAD * NPAD * 4 + NPAD * NPAD * 2 + 3584; // 249,344
  const int MLP_SMEM  = 384 * CH * 2 * 2 + 128 * 384 * 2 + 384 * 4 + CH * 4;          // 247,680
  (void)hipFuncSetAttribute((const void*)k_attn, hipFuncAttributeMaxDynamicSharedMemorySize, ATTN_SMEM);
  (void)hipFuncSetAttribute((const void*)k_mlp,  hipFuncAttributeMaxDynamicSharedMemorySize, MLP_SMEM);

  k_ln_partition<<<TPAD / 256, 256, 0, stream>>>(Im, If, ln1q_g, ln1q_b, ln1kv_g, ln1kv_b, xw, yw);
  k_gemm_f16<<<TPAD / 128, 256, 0, stream>>>(xw, wq, bq, Qb, Qb, 96);
  k_gemm_f16<<<TPAD / 128, 256, 0, stream>>>(yw, wkv, bkv, Kb, Vb, 192);
  k_attn<<<NWIN, 256, ATTN_SMEM, stream>>>(Qb, Kb, Vb, relb, Ob);
  k_gemm_proj<<<TPAD / 128, 256, 0, stream>>>(Ob, wproj, bproj, PR);
  k_residual_ln<<<NTOT / 256, 256, 0, stream>>>(Im, PR, ln2_g, ln2_b, Xb, hb);
  k_mlp<<<NTOT / 128, 256, MLP_SMEM, stream>>>(hb, Xb, mlp_w1, mlp_b1, mlp_w2, mlp_b2, x2);
  k_conv<<<460800 / 256, 256, 0, stream>>>(x2, conv_w, conv_b, (float*)d_out);
}
